// DecoderWithAttention_choice_41961830482518
// MI455X (gfx1250) — matrix operand order repaired and checked by reading, not compiled, not run on hardware
//
#include <hip/hip_runtime.h>
#include <hip/hip_bf16.h>
#include <math.h>

// Problem dims
#define B_   128
#define P_   64
#define ENC_ 2048
#define E_   512
#define D_   512
#define V_   10000
#define T_   32
#define TDEC 31

typedef __attribute__((ext_vector_type(16))) __bf16 v16bf;
typedef __attribute__((ext_vector_type(8)))  __bf16 v8bf;
typedef __attribute__((ext_vector_type(8)))  float  v8f;

__device__ __forceinline__ __bf16 f2bf(float f) {
    unsigned u = __builtin_bit_cast(unsigned, f);
    unsigned r = u + 0x7FFFu + ((u >> 16) & 1u);   // round-to-nearest-even
    unsigned short h = (unsigned short)(r >> 16);
    return __builtin_bit_cast(__bf16, h);
}
__device__ __forceinline__ float sigf(float x) { return 1.0f / (1.0f + __expf(-x)); }

// ---------------------------------------------------------------------------
// Batch sort by caption length (descending, stable) + int outputs
// ---------------------------------------------------------------------------
__global__ void sort_k(const int* __restrict__ lengths, const int* __restrict__ caps,
                       int* __restrict__ order, int* __restrict__ dec_len,
                       int* __restrict__ out_caps, int* __restrict__ out_dec,
                       int* __restrict__ out_order) {
    __shared__ int sl[B_], so[B_];
    int tid = threadIdx.x;
    int li = lengths[tid];
    sl[tid] = li;
    __syncthreads();
    int rank = 0;
    for (int j = 0; j < B_; j++) {
        int lj = sl[j];
        rank += (lj > li) || (lj == li && j < tid);
    }
    so[rank] = tid;
    __syncthreads();
    int o = so[tid];
    order[tid] = o;
    int dl = sl[o] - 1;
    dec_len[tid] = dl;
    out_order[tid] = o;
    out_dec[tid] = dl;
    for (int t = 0; t < T_; t++) out_caps[tid * T_ + t] = caps[o * T_ + t];
}

// ---------------------------------------------------------------------------
// f32 -> bf16 convert
// ---------------------------------------------------------------------------
__global__ void cvt_k(const float* __restrict__ src, __bf16* __restrict__ dst, int n) {
    int i = blockIdx.x * blockDim.x + threadIdx.x;
    if (i < n) dst[i] = f2bf(src[i]);
}

__global__ void zero_k(unsigned* __restrict__ p, int n) {
    int i = blockIdx.x * blockDim.x + threadIdx.x;
    if (i < n) p[i] = 0u;
}

// ---------------------------------------------------------------------------
// mean over P pixels of gathered encoder rows: me[b][k]
// ---------------------------------------------------------------------------
__global__ void meanenc_k(const float* __restrict__ enc, const int* __restrict__ order,
                          float* __restrict__ me) {
    int idx = blockIdx.x * blockDim.x + threadIdx.x;
    if (idx >= B_ * ENC_) return;
    int b = idx >> 11, k = idx & (ENC_ - 1);
    const float* base = enc + ((size_t)order[b] * P_) * ENC_ + k;
    float s = 0.f;
    for (int p = 0; p < P_; p++) s += base[(size_t)p * ENC_];
    me[idx] = s * (1.0f / P_);
}

// global_img = me @ W_glob^T + b_glob   (small, f32 for accuracy)
__global__ void gimg_k(const float* __restrict__ me, const float* __restrict__ Wg,
                       const float* __restrict__ bg, float* __restrict__ gimg) {
    int idx = blockIdx.x * blockDim.x + threadIdx.x;
    if (idx >= B_ * D_) return;
    int b = idx >> 9, d = idx & (D_ - 1);
    const float* m = me + (size_t)b * ENC_;
    const float* wrow = Wg + (size_t)d * ENC_;
    float s = bg[d];
    for (int k = 0; k < ENC_; k++) s += m[k] * wrow[k];
    gimg[idx] = s;
}

// ---------------------------------------------------------------------------
// Fragment loaders for the CDNA5 16x16x32 bf16 WMMA VGPR layouts.
// A (16x32, MxK): lane(0..15)=M, lane half selects K sub-octet; row-major K-contig.
// B (32x16, KxN): lane(0..15)=N, lane half selects K sub-octet (same as A); W stored (N,K).
// ---------------------------------------------------------------------------
__device__ __forceinline__ v16bf load_a_bf(const __bf16* __restrict__ ar, int kb8) {
    v8bf lo = *(const v8bf*)(ar + kb8);
    v8bf hi = *(const v8bf*)(ar + 16 + kb8);
    v16bf a;
#pragma unroll
    for (int i = 0; i < 8; i++) { a[i] = lo[i]; a[i + 8] = hi[i]; }
    return a;
}
__device__ __forceinline__ v16bf load_a_f32(const float* __restrict__ ar, int kb8) {
    v16bf a;
#pragma unroll
    for (int i = 0; i < 8; i++) {
        a[i]     = f2bf(ar[kb8 + i]);
        a[i + 8] = f2bf(ar[16 + kb8 + i]);
    }
    return a;
}
__device__ __forceinline__ v16bf load_b_frag(const __bf16* __restrict__ bp) {
    v8bf b0 = *(const v8bf*)(bp);
    v8bf b1 = *(const v8bf*)(bp + 16);
    v16bf bb;
#pragma unroll
    for (int i = 0; i < 8; i++) { bb[i] = b0[i]; bb[i + 8] = b1[i]; }
    return bb;
}

// ---------------------------------------------------------------------------
// Generic bf16 WMMA GEMM: C[M,N] (+)= A[M,K] * W[N,K]^T  (+bias) (mask rows)
// Wave computes a 32x64 tile: 2 A frags x 4 B frags -> 8 WMMAs per 32-K step.
// Explicit 2-stage ping-pong software pipeline over 64-wide K chunks:
//  - distinct stage-0/stage-1 fragment variables -> no register rotation copies
//  - loads target the idle stage's registers -> no WMMA WAR hazard v_nops
//  - affine offsets (k0+32, k0+64) -> immediate-offset address folding
// Requires: M % 32 == 0, K % 64 == 0, K >= 128 (true for all uses here).
// AF32: A is f32, converted to bf16 in-flight; optional sorted-batch row gather.
// ---------------------------------------------------------------------------
template <bool AF32>
__global__ __launch_bounds__(128) void gemm_wmma_k(
    const void* __restrict__ Ap, int lda,
    const __bf16* __restrict__ Bw, int ldb,
    float* __restrict__ C, int ldc,
    int M, int N, int K, int betaAdd,
    const float* __restrict__ bias,
    const int* __restrict__ mask_len, int mask_t,
    const int* __restrict__ rowmap, int rowP) {
    const int tilesN = (N + 63) >> 6;
    const int tilesM = M >> 5;
    const int wave = (int)((blockIdx.x * blockDim.x + threadIdx.x) >> 5);
    if (wave >= tilesM * tilesN) return;   // whole-wave exit: EXEC stays all-1 for WMMA
    const int lane = threadIdx.x & 31;
    const int tm = wave / tilesN;
    const int tn = wave % tilesN;
    const int half = lane >> 4;
    const int l15  = lane & 15;

    // Two A fragment rows (per-lane M), optional gather through sort order
    int arow0 = tm * 32 + l15;
    int arow1 = arow0 + 16;
    if (AF32 && rowmap) {
        arow0 = rowmap[arow0 / rowP] * rowP + (arow0 % rowP);
        arow1 = rowmap[arow1 / rowP] * rowP + (arow1 % rowP);
    }

    const int kb8  = half * 8;    // A: K = kb8..kb8+7 and 16+kb8..16+kb8+7
    const int kb16 = half * 8;    // B: K = kb16..kb16+7 and 16+kb16..16+kb16+7 for column l15

    // B fragment base pointers, clamped at N edge (results discarded at store)
    const __bf16* Bp[4];
#pragma unroll
    for (int s4 = 0; s4 < 4; s4++) {
        int col = tn * 64 + s4 * 16 + l15;
        if (col > N - 1) col = N - 1;
        Bp[s4] = Bw + (size_t)col * ldb + kb16;
    }

    const __bf16* Abf = (const __bf16*)Ap;
    const float*  Afp = (const float*)Ap;
    const __bf16* A0b = Abf + (size_t)arow0 * lda;
    const __bf16* A1b = Abf + (size_t)arow1 * lda;
    const float*  A0f = Afp + (size_t)arow0 * lda;
    const float*  A1f = Afp + (size_t)arow1 * lda;

    v8f acc[2][4] = {};

    // Stage-0 fragments for k = 0
    v16bf a0_0 = AF32 ? load_a_f32(A0f, kb8) : load_a_bf(A0b, kb8);
    v16bf a1_0 = AF32 ? load_a_f32(A1f, kb8) : load_a_bf(A1b, kb8);
    v16bf b_0[4];
#pragma unroll
    for (int s4 = 0; s4 < 4; s4++) b_0[s4] = load_b_frag(Bp[s4]);

    v16bf a0_1, a1_1, b_1[4];

    int k0 = 0;
    for (; k0 < K - 64; k0 += 64) {
        // load stage-1 fragments (k0+32)
        a0_1 = AF32 ? load_a_f32(A0f + k0 + 32, kb8) : load_a_bf(A0b + k0 + 32, kb8);
        a1_1 = AF32 ? load_a_f32(A1f + k0 + 32, kb8) : load_a_bf(A1b + k0 + 32, kb8);
#pragma unroll
        for (int s4 = 0; s4 < 4; s4++) b_1[s4] = load_b_frag(Bp[s4] + k0 + 32);
        // compute stage 0 (k0)
#pragma unroll
        for (int s4 = 0; s4 < 4; s4++) {
            acc[0][s4] = __builtin_amdgcn_wmma_f32_16x16x32_bf16(
                false, a0_0, false, b_0[s4], (short)0, acc[0][s4], false, false);
            acc[1][s4] = __builtin_amdgcn_wmma_f32_16x16x32_bf16(
                false, a1_0, false, b_0[s4], (short)0, acc[1][s4], false, false);
        }
        // load stage-0 fragments (k0+64)
        a0_0 = AF32 ? load_a_f32(A0f + k0 + 64, kb8) : load_a_bf(A0b + k0 + 64, kb8);
        a1_0 = AF32 ? load_a_f32(A1f + k0 + 64, kb8) : load_a_bf(A1b + k0 + 64, kb8);
#pragma unroll
        for (int s4 = 0; s4 < 4; s4++) b_0[s4] = load_b_frag(Bp[s4] + k0 + 64);
        // compute stage 1 (k0+32)
#pragma unroll
        for (int s4 = 0; s4 < 4; s4++) {
            acc[0][s4] = __builtin_amdgcn_wmma_f32_16x16x32_bf16(
                false, a0_1, false, b_1[s4], (short)0, acc[0][s4], false, false);
            acc[1][s4] = __builtin_amdgcn_wmma_f32_16x16x32_bf16(
                false, a1_1, false, b_1[s4], (short)0, acc[1][s4], false, false);
        }
    }
    // Epilogue chunk (k0 == K-64): stage-0 frags already loaded
    a0_1 = AF32 ? load_a_f32(A0f + k0 + 32, kb8) : load_a_bf(A0b + k0 + 32, kb8);
    a1_1 = AF32 ? load_a_f32(A1f + k0 + 32, kb8) : load_a_bf(A1b + k0 + 32, kb8);
#pragma unroll
    for (int s4 = 0; s4 < 4; s4++) b_1[s4] = load_b_frag(Bp[s4] + k0 + 32);
#pragma unroll
    for (int s4 = 0; s4 < 4; s4++) {
        acc[0][s4] = __builtin_amdgcn_wmma_f32_16x16x32_bf16(
            false, a0_0, false, b_0[s4], (short)0, acc[0][s4], false, false);
        acc[1][s4] = __builtin_amdgcn_wmma_f32_16x16x32_bf16(
            false, a1_0, false, b_0[s4], (short)0, acc[1][s4], false, false);
    }
#pragma unroll
    for (int s4 = 0; s4 < 4; s4++) {
        acc[0][s4] = __builtin_amdgcn_wmma_f32_16x16x32_bf16(
            false, a0_1, false, b_1[s4], (short)0, acc[0][s4], false, false);
        acc[1][s4] = __builtin_amdgcn_wmma_f32_16x16x32_bf16(
            false, a1_1, false, b_1[s4], (short)0, acc[1][s4], false, false);
    }

    // Store: C layout lane(l15)=N, vgpr r -> M = r + 8*half (+16 for second frag)
#pragma unroll
    for (int s4 = 0; s4 < 4; s4++) {
        int cc = tn * 64 + s4 * 16 + l15;
        if (cc >= N) continue;
#pragma unroll
        for (int mi = 0; mi < 2; mi++) {
#pragma unroll
            for (int r = 0; r < 8; r++) {
                int m = tm * 32 + mi * 16 + half * 8 + r;
                float v = acc[mi][s4][r];
                if (betaAdd) v += C[(size_t)m * ldc + cc];
                if (bias) v += bias[cc];
                if (mask_len && mask_len[m] <= mask_t) v = 0.0f;   // inactive step
                C[(size_t)m * ldc + cc] = v;
            }
        }
    }
}

// ---------------------------------------------------------------------------
// Per-step input build: x = [embedding(token_t), global_img] -> bf16
// ---------------------------------------------------------------------------
__global__ void prep_x_k(const float* __restrict__ emb_table, const int* __restrict__ caps,
                         const int* __restrict__ order, const float* __restrict__ gimg,
                         int t, __bf16* __restrict__ xbf) {
    int idx = blockIdx.x * blockDim.x + threadIdx.x;
    if (idx >= B_ * (E_ + D_)) return;
    int b = idx >> 10, j = idx & 1023;
    float v;
    if (j < E_) {
        int tok = caps[order[b] * T_ + t];
        v = emb_table[(size_t)tok * E_ + j];
    } else {
        v = gimg[b * D_ + (j - E_)];
    }
    xbf[idx] = f2bf(v);
}

// ---------------------------------------------------------------------------
// LSTM cell + sentinel gate (f32 for recurrence accuracy)
// ---------------------------------------------------------------------------
__global__ void cell_k(const float* __restrict__ gates, const float* __restrict__ gx,
                       const float* __restrict__ b_ih, const float* __restrict__ b_hh,
                       float* __restrict__ h, float* __restrict__ c, float* __restrict__ s_t,
                       __bf16* __restrict__ h_bf, __bf16* __restrict__ st_bf) {
    int idx = blockIdx.x * blockDim.x + threadIdx.x;
    if (idx >= B_ * D_) return;
    int b = idx >> 9, d = idx & (D_ - 1);
    const float* g = gates + (size_t)b * (4 * D_);
    float i_ = g[d]            + b_ih[d]            + b_hh[d];
    float f_ = g[D_ + d]       + b_ih[D_ + d]       + b_hh[D_ + d];
    float g_ = g[2 * D_ + d]   + b_ih[2 * D_ + d]   + b_hh[2 * D_ + d];
    float o_ = g[3 * D_ + d]   + b_ih[3 * D_ + d]   + b_hh[3 * D_ + d];
    float cn = sigf(f_) * c[idx] + sigf(i_) * tanhf(g_);
    float tc = tanhf(cn);
    float hn = sigf(o_) * tc;
    float st = sigf(gx[idx]) * tc;
    c[idx] = cn; h[idx] = hn; s_t[idx] = st;
    h_bf[idx] = f2bf(hn); st_bf[idx] = f2bf(st);
}

// ---------------------------------------------------------------------------
// Fused adaptive attention: z, z_ex, softmaxes, beta, context, fcin=bf16(h+ctx')
// One block per batch row; P=64 fits in LDS.
// ---------------------------------------------------------------------------
__global__ __launch_bounds__(256) void attention_k(
    const float* __restrict__ enc_v, const float* __restrict__ enc_small,
    const float* __restrict__ hg, const float* __restrict__ sv,
    const float* __restrict__ W_ht, const float* __restrict__ s_t,
    const float* __restrict__ h_new, __bf16* __restrict__ fcin) {
    int b = blockIdx.x, tid = threadIdx.x;
    __shared__ float shg[P_], swht[P_], zsh[P_], salpha[P_];
    __shared__ float sbeta;
    if (tid < P_) { shg[tid] = hg[b * P_ + tid]; swht[tid] = W_ht[tid]; }
    __syncthreads();
    if (tid < P_) {
        const float* ev = enc_v + ((size_t)b * P_ + tid) * P_;
        float acc = 0.f;
#pragma unroll 8
        for (int q = 0; q < P_; q++) acc += tanhf(ev[q] + shg[q]) * swht[q];
        zsh[tid] = acc;
        salpha[tid] = tanhf(sv[b * P_ + tid] + shg[tid]) * swht[tid];  // z_ex partials
    }
    __syncthreads();
    if (tid == 0) {
        float zex = 0.f, mz = -1e30f;
        for (int p = 0; p < P_; p++) { zex += salpha[p]; if (zsh[p] > mz) mz = zsh[p]; }
        float sez = 0.f;
        for (int p = 0; p < P_; p++) { float e = __expf(zsh[p] - mz); salpha[p] = e; sez += e; }
        float inv = 1.f / sez;
        for (int p = 0; p < P_; p++) salpha[p] *= inv;                 // alpha
        float m2 = mz > zex ? mz : zex;
        float d2 = sez * __expf(mz - m2) + __expf(zex - m2);
        sbeta = __expf(zex - m2) / d2;                                 // beta
    }
    __syncthreads();
    float beta = sbeta;
    for (int d = tid; d < D_; d += 256) {
        float cacc = 0.f;
        const float* es = enc_small + ((size_t)b * P_) * D_ + d;
#pragma unroll 4
        for (int p = 0; p < P_; p++) cacc += salpha[p] * es[(size_t)p * D_];
        float ctxp = s_t[b * D_ + d] * beta + cacc * (1.f - beta);
        fcin[b * D_ + d] = f2bf(h_new[b * D_ + d] + ctxp);
    }
}

// ---------------------------------------------------------------------------
// Host-side launch helpers (M multiple of 32, K multiple of 64 — true here)
// ---------------------------------------------------------------------------
static void launch_gemm(const __bf16* A, int lda, const __bf16* Bw, int ldb,
                        float* C, int ldc, int M, int N, int K, int betaAdd,
                        const float* bias, const int* mask, int mt, hipStream_t s) {
    int waves = (M / 32) * ((N + 63) / 64);
    int blocks = (waves * 32 + 127) / 128;
    gemm_wmma_k<false><<<blocks, 128, 0, s>>>((const void*)A, lda, Bw, ldb, C, ldc,
                                              M, N, K, betaAdd, bias, mask, mt, nullptr, 0);
}
static void launch_gemm_f32A(const float* A, int lda, const __bf16* Bw, int ldb,
                             float* C, int ldc, int M, int N, int K,
                             const int* rowmap, int rowP, hipStream_t s) {
    int waves = (M / 32) * ((N + 63) / 64);
    int blocks = (waves * 32 + 127) / 128;
    gemm_wmma_k<true><<<blocks, 128, 0, s>>>((const void*)A, lda, Bw, ldb, C, ldc,
                                             M, N, K, 0, nullptr, nullptr, 0, rowmap, rowP);
}

extern "C" void kernel_launch(void* const* d_in, const int* in_sizes, int n_in,
                              void* d_out, int out_size, void* d_ws, size_t ws_size,
                              hipStream_t stream) {
    (void)in_sizes; (void)n_in; (void)out_size; (void)ws_size;
    const float* enc     = (const float*)d_in[0];
    const int*   caps    = (const int*)d_in[1];
    const int*   lens    = (const int*)d_in[2];
    const float* embt    = (const float*)d_in[3];
    const float* W_ih    = (const float*)d_in[4];
    const float* W_hh    = (const float*)d_in[5];
    const float* b_ih    = (const float*)d_in[6];
    const float* b_hh    = (const float*)d_in[7];
    const float* W_glob  = (const float*)d_in[8];
    const float* b_glob  = (const float*)d_in[9];
    const float* W_small = (const float*)d_in[10];
    const float* W_v     = (const float*)d_in[11];
    const float* W_g     = (const float*)d_in[12];
    const float* W_ht    = (const float*)d_in[13];
    const float* W_x     = (const float*)d_in[14];
    const float* W_h     = (const float*)d_in[15];
    const float* W_s     = (const float*)d_in[16];
    const float* W_fc    = (const float*)d_in[17];
    const float* b_fc    = (const float*)d_in[18];

    // Output layout: predictions(B,TDEC,V) f32 | caps(B,T) i32 | dec_len(B) i32 | order(B) i32
    float* out_pred = (float*)d_out;
    int* out_caps  = (int*)((char*)d_out + (size_t)B_ * TDEC * V_ * 4);
    int* out_dec   = out_caps + B_ * T_;
    int* out_order = out_dec + B_;

    // Workspace carve-out (~51 MB)
    char* w = (char*)d_ws;
    size_t off = 0;
    auto alloc = [&](size_t bytes) -> char* {
        off = (off + 255) & ~(size_t)255;
        char* p = w + off;
        off += bytes;
        return p;
    };
    int*    order     = (int*)alloc(B_ * 4);
    int*    dec_len   = (int*)alloc(B_ * 4);
    __bf16* Wsm_bf    = (__bf16*)alloc((size_t)D_ * ENC_ * 2);
    __bf16* Wv_bf     = (__bf16*)alloc((size_t)P_ * D_ * 2);
    __bf16* Wih_bf    = (__bf16*)alloc((size_t)4 * D_ * (E_ + D_) * 2);
    __bf16* Whh_bf    = (__bf16*)alloc((size_t)4 * D_ * D_ * 2);
    __bf16* Wx_bf     = (__bf16*)alloc((size_t)D_ * 2 * D_ * 2);
    __bf16* Wh_bf     = (__bf16*)alloc((size_t)D_ * D_ * 2);
    __bf16* Wg_bf     = (__bf16*)alloc((size_t)P_ * D_ * 2);
    __bf16* Ws_bf     = (__bf16*)alloc((size_t)P_ * D_ * 2);
    __bf16* Wfc_bf    = (__bf16*)alloc((size_t)V_ * D_ * 2);
    float*  me        = (float*)alloc((size_t)B_ * ENC_ * 4);
    float*  gimg      = (float*)alloc((size_t)B_ * D_ * 4);
    float*  enc_sm    = (float*)alloc((size_t)B_ * P_ * D_ * 4);
    __bf16* enc_sm_bf = (__bf16*)alloc((size_t)B_ * P_ * D_ * 2);
    float*  enc_v     = (float*)alloc((size_t)B_ * P_ * P_ * 4);
    float*  gates     = (float*)alloc((size_t)B_ * 4 * D_ * 4);
    float*  gx        = (float*)alloc((size_t)B_ * D_ * 4);
    float*  hbuf      = (float*)alloc((size_t)B_ * D_ * 4);
    float*  cbuf      = (float*)alloc((size_t)B_ * D_ * 4);
    float*  stbuf     = (float*)alloc((size_t)B_ * D_ * 4);
    float*  hg        = (float*)alloc((size_t)B_ * P_ * 4);
    float*  sv        = (float*)alloc((size_t)B_ * P_ * 4);
    __bf16* xbf       = (__bf16*)alloc((size_t)B_ * (E_ + D_) * 2);
    __bf16* hbf       = (__bf16*)alloc((size_t)B_ * D_ * 2);
    __bf16* stbf      = (__bf16*)alloc((size_t)B_ * D_ * 2);
    __bf16* fcin      = (__bf16*)alloc((size_t)B_ * D_ * 2);

    // 1) Sort + int outputs
    sort_k<<<1, B_, 0, stream>>>(lens, caps, order, dec_len, out_caps, out_dec, out_order);

    // 2) Weight conversions to bf16 (L2-resident afterwards)
    auto cvt = [&](const float* s, __bf16* d, int n) {
        cvt_k<<<(n + 255) / 256, 256, 0, stream>>>(s, d, n);
    };
    cvt(W_small, Wsm_bf, D_ * ENC_);
    cvt(W_v,     Wv_bf,  P_ * D_);
    cvt(W_ih,    Wih_bf, 4 * D_ * (E_ + D_));
    cvt(W_hh,    Whh_bf, 4 * D_ * D_);
    cvt(W_x,     Wx_bf,  D_ * 2 * D_);
    cvt(W_h,     Wh_bf,  D_ * D_);
    cvt(W_g,     Wg_bf,  P_ * D_);
    cvt(W_s,     Ws_bf,  P_ * D_);
    cvt(W_fc,    Wfc_bf, V_ * D_);

    // 3) global_img (f32)
    meanenc_k<<<(B_ * ENC_ + 255) / 256, 256, 0, stream>>>(enc, order, me);
    gimg_k<<<(B_ * D_ + 255) / 256, 256, 0, stream>>>(me, W_glob, b_glob, gimg);

    // 4) enc_small = gather(enc)@W_small^T  (WMMA, f32 A with in-flight bf16 + row gather)
    launch_gemm_f32A(enc, ENC_, Wsm_bf, ENC_, enc_sm, D_, B_ * P_, D_, ENC_, order, P_, stream);
    cvt(enc_sm, enc_sm_bf, B_ * P_ * D_);
    // enc_v = enc_small @ W_v^T
    launch_gemm(enc_sm_bf, D_, Wv_bf, D_, enc_v, P_, B_ * P_, P_, D_, 0, nullptr, nullptr, 0, stream);

    // 5) init recurrent state
    zero_k<<<(B_ * D_ / 2 + 255) / 256, 256, 0, stream>>>((unsigned*)hbf, B_ * D_ / 2);
    zero_k<<<(B_ * D_ + 255) / 256, 256, 0, stream>>>((unsigned*)cbuf, B_ * D_);

    // 6) decode loop
    for (int t = 0; t < TDEC; t++) {
        prep_x_k<<<(B_ * (E_ + D_) + 255) / 256, 256, 0, stream>>>(embt, caps, order, gimg, t, xbf);
        // gates = x@W_ih^T + h@W_hh^T
        launch_gemm(xbf, E_ + D_, Wih_bf, E_ + D_, gates, 4 * D_, B_, 4 * D_, E_ + D_, 0, nullptr, nullptr, 0, stream);
        launch_gemm(hbf, D_,      Whh_bf, D_,      gates, 4 * D_, B_, 4 * D_, D_,      1, nullptr, nullptr, 0, stream);
        // gx = x@W_x^T + h@W_h^T
        launch_gemm(xbf, E_ + D_, Wx_bf, E_ + D_, gx, D_, B_, D_, E_ + D_, 0, nullptr, nullptr, 0, stream);
        launch_gemm(hbf, D_,      Wh_bf, D_,      gx, D_, B_, D_, D_,      1, nullptr, nullptr, 0, stream);
        // LSTM cell + sentinel gate (updates hbf in place for next step's gate GEMMs)
        cell_k<<<(B_ * D_ + 255) / 256, 256, 0, stream>>>(gates, gx, b_ih, b_hh,
                                                          hbuf, cbuf, stbuf, hbf, stbf);
        // hg = h_new@W_g^T ; sv = s_t@W_s^T
        launch_gemm(hbf,  D_, Wg_bf, D_, hg, P_, B_, P_, D_, 0, nullptr, nullptr, 0, stream);
        launch_gemm(stbf, D_, Ws_bf, D_, sv, P_, B_, P_, D_, 0, nullptr, nullptr, 0, stream);
        // fused attention -> fcin
        attention_k<<<B_, 256, 0, stream>>>(enc_v, enc_sm, hg, sv, W_ht, stbuf, hbuf, fcin);
        // preds[:, t, :] = fcin@W_fc^T + b_fc, masked by activity, written into d_out
        launch_gemm(fcin, D_, Wfc_bf, D_, out_pred + (size_t)t * V_, TDEC * V_,
                    B_, V_, D_, 0, b_fc, dec_len, t, stream);
    }
}
